// CrossAttention_21345987461158
// MI455X (gfx1250) — matrix operand order repaired and checked by reading, not compiled, not run on hardware
//
#include <hip/hip_runtime.h>

typedef _Float16 half16 __attribute__((ext_vector_type(16)));
typedef _Float16 half8  __attribute__((ext_vector_type(8)));
typedef float    float8 __attribute__((ext_vector_type(8)));

#define HEADS     8
#define DIM_HEAD  64
#define INNER     512
#define QDIM      512
#define CDIM      768
#define BATCH     4
#define SEQ       2048
#define SCALE_F   0.125f   // 64^-0.5

#define KT_LD     80       // padded LDS row stride (halfs) for K tile rows (160 B)
#define VT_LD     40       // padded LDS row stride (halfs) for transposed V rows (80 B)

static __device__ inline float8 wmma_f16(half16 a, half16 b, float8 c) {
    return __builtin_amdgcn_wmma_f32_16x16x32_f16(false, a, false, b, (short)0, c,
                                                  false, false);
}

// ---- Fragment loaders (16x16x32 f16 WMMA, wave32; contiguous k order, A and B alike) -----------------
// A (16x32, MxK): lane L holds row M = L&15; half h = L>>4.
//   element (g*8+j) = A[row][k0 + h*16 + g*8 + j]
static __device__ inline half16 load_a_f32(const float* __restrict__ A, int lda,
                                           int row0, int k0, int lane) {
    const float* p = A + (size_t)(row0 + (lane & 15)) * lda + k0 + ((lane >> 4) << 4);
    half16 a;
#pragma unroll
    for (int g = 0; g < 2; ++g)
#pragma unroll
        for (int j = 0; j < 8; ++j)
            a[g * 8 + j] = (_Float16)p[g * 8 + j];
    return a;
}

static __device__ inline half16 load_a_f16(const _Float16* __restrict__ A, int lda,
                                           int row0, int k0, int lane) {
    const _Float16* p = A + (size_t)(row0 + (lane & 15)) * lda + k0 + ((lane >> 4) << 4);
    half16 a;
#pragma unroll
    for (int g = 0; g < 2; ++g)
#pragma unroll
        for (int j = 0; j < 8; ++j)
            a[g * 8 + j] = p[g * 8 + j];
    return a;
}

// B (32x16, KxN): lane L holds column N = L&15; half h = L>>4.
//   element j = B[k0 + h*16 + j][col];  source holds B^T rows (16 contiguous halfs/lane)
static __device__ inline half16 load_b_tr(const _Float16* __restrict__ W, int ldw,
                                          int col0, int k0, int lane) {
    const _Float16* p = W + (size_t)(col0 + (lane & 15)) * ldw + k0 + ((lane >> 4) << 4);
    half16 b;
#pragma unroll
    for (int j = 0; j < 16; ++j) b[j] = p[j];
    return b;
}

// Same pattern but for LDS tiles (small strides, int math)
static __device__ inline half16 lds_b_frag(const _Float16* base, int row0, int ld,
                                           int k0, int lane) {
    const _Float16* p = base + (row0 + (lane & 15)) * ld + k0 + ((lane >> 4) << 4);
    half16 b;
#pragma unroll
    for (int j = 0; j < 16; ++j) b[j] = p[j];
    return b;
}

// ---- f32 -> f16 convert ----------------------------------------------------
__global__ void cvt_f16(const float* __restrict__ src, _Float16* __restrict__ dst, int n) {
    int i = blockIdx.x * blockDim.x + threadIdx.x;
    if (i < n) dst[i] = (_Float16)src[i];
}

// ---- GEMM: C_f16[M,N] = A_f32[M,K] @ W_f16[N,K]^T --------------------------
// block = 128 threads (4 waves). wave w -> 32x32 tile.
__global__ void gemm_xw_f16(const float* __restrict__ A, const _Float16* __restrict__ W,
                            _Float16* __restrict__ C, int M, int K, int N) {
    int lane = threadIdx.x & 31;
    int wave = threadIdx.x >> 5;
    int rb = blockIdx.x * 32;
    int cb = (blockIdx.y * 4 + wave) * 32;
    float8 acc00 = {}, acc01 = {}, acc10 = {}, acc11 = {};
    for (int kk = 0; kk < K; kk += 32) {
        half16 a0 = load_a_f32(A, K, rb, kk, lane);
        half16 a1 = load_a_f32(A, K, rb + 16, kk, lane);
        half16 b0 = load_b_tr(W, K, cb, kk, lane);
        half16 b1 = load_b_tr(W, K, cb + 16, kk, lane);
        acc00 = wmma_f16(a0, b0, acc00);
        acc01 = wmma_f16(a0, b1, acc01);
        acc10 = wmma_f16(a1, b0, acc10);
        acc11 = wmma_f16(a1, b1, acc11);
    }
    int rr = (lane >> 4) << 3;
    int cc = lane & 15;
#pragma unroll
    for (int i = 0; i < 8; ++i) {
        C[(size_t)(rb + rr + i) * N + cb + cc]           = (_Float16)acc00[i];
        C[(size_t)(rb + rr + i) * N + cb + 16 + cc]      = (_Float16)acc01[i];
        C[(size_t)(rb + 16 + rr + i) * N + cb + cc]      = (_Float16)acc10[i];
        C[(size_t)(rb + 16 + rr + i) * N + cb + 16 + cc] = (_Float16)acc11[i];
    }
}

// ---- GEMM: out_f32[M,N] = A_f16[M,K] @ W_f16[N,K]^T + bias -----------------
__global__ void gemm_ow_f32(const _Float16* __restrict__ A, const _Float16* __restrict__ W,
                            const float* __restrict__ bias, float* __restrict__ C,
                            int M, int K, int N) {
    int lane = threadIdx.x & 31;
    int wave = threadIdx.x >> 5;
    int rb = blockIdx.x * 32;
    int cb = (blockIdx.y * 4 + wave) * 32;
    float8 acc00 = {}, acc01 = {}, acc10 = {}, acc11 = {};
    for (int kk = 0; kk < K; kk += 32) {
        half16 a0 = load_a_f16(A, K, rb, kk, lane);
        half16 a1 = load_a_f16(A, K, rb + 16, kk, lane);
        half16 b0 = load_b_tr(W, K, cb, kk, lane);
        half16 b1 = load_b_tr(W, K, cb + 16, kk, lane);
        acc00 = wmma_f16(a0, b0, acc00);
        acc01 = wmma_f16(a0, b1, acc01);
        acc10 = wmma_f16(a1, b0, acc10);
        acc11 = wmma_f16(a1, b1, acc11);
    }
    int rr = (lane >> 4) << 3;
    int cc = lane & 15;
    float bc0 = bias[cb + cc];
    float bc1 = bias[cb + 16 + cc];
#pragma unroll
    for (int i = 0; i < 8; ++i) {
        C[(size_t)(rb + rr + i) * N + cb + cc]           = acc00[i] + bc0;
        C[(size_t)(rb + rr + i) * N + cb + 16 + cc]      = acc01[i] + bc1;
        C[(size_t)(rb + 16 + rr + i) * N + cb + cc]      = acc10[i] + bc0;
        C[(size_t)(rb + 16 + rr + i) * N + cb + 16 + cc] = acc11[i] + bc1;
    }
}

// ---- Fused flash attention -------------------------------------------------
// grid = (SEQ/128, HEADS, BATCH), block = 256 (8 waves); wave w -> 16 query rows.
// K tile staged via gfx1250 async global->LDS DMA (ASYNCcnt); V staged transposed.
__global__ void attn_flash(const _Float16* __restrict__ Q, const _Float16* __restrict__ K,
                           const _Float16* __restrict__ V, _Float16* __restrict__ O) {
    __shared__ _Float16 Kt[32][KT_LD];       // K tile: [ctx_row][d]
    __shared__ _Float16 Vt[64][VT_LD];       // V tile transposed: [d][ctx_row]
    __shared__ _Float16 pbuf[8][16 * 32];    // per-wave P staging

    int tid  = threadIdx.x;
    int lane = tid & 31;
    int wave = tid >> 5;
    int b    = blockIdx.z;
    int head = blockIdx.y;
    int qb   = blockIdx.x * 128 + wave * 16;

    const _Float16* Qb = Q + (size_t)b * SEQ * INNER + head * DIM_HEAD;
    const _Float16* Kb = K + (size_t)b * SEQ * INNER + head * DIM_HEAD;
    const _Float16* Vb = V + (size_t)b * SEQ * INNER + head * DIM_HEAD;
    _Float16*       Ob = O + (size_t)b * SEQ * INNER + head * DIM_HEAD;

    // Q fragments for this 16-row tile (d = 0..31 and 32..63), loaded once
    half16 qa0 = load_a_f16(Qb, INNER, qb, 0, lane);
    half16 qa1 = load_a_f16(Qb, INNER, qb, 32, lane);

    float m_i[8], l_i[8];
#pragma unroll
    for (int i = 0; i < 8; ++i) { m_i[i] = -1e30f; l_i[i] = 0.0f; }
    float8 o0 = {}, o1 = {}, o2 = {}, o3 = {};

    _Float16* pb = &pbuf[wave][0];
    int rr = (lane >> 4) << 3;
    int cc = lane & 15;
    float8 z = {};

    // staging coordinates: 256 threads cover 32 rows x 8 chunks of 8 halfs
    int srow = tid >> 3;
    int scol = (tid & 7) << 3;
    // per-lane LDS byte address for the async K-tile DMA (padding preserved)
    unsigned klds = (unsigned)(uintptr_t)&Kt[srow][scol];

    for (int jt = 0; jt < SEQ; jt += 32) {
        __syncthreads();   // previous tile fully consumed

        // ---- K tile: async global->LDS (no VGPR round-trip, ASYNCcnt) ----
        {
            const _Float16* ksrc = Kb + (size_t)(jt + srow) * INNER + scol;
            asm volatile("global_load_async_to_lds_b128 %0, %1, off"
                         :: "v"(klds), "v"(ksrc)
                         : "memory");
        }
        // ---- V tile staged transposed (DMA cannot transpose) ----
        {
            const _Float16* vsrc = Vb + (size_t)(jt + srow) * INNER + scol;
            half8 vv = *(const half8*)vsrc;
#pragma unroll
            for (int j = 0; j < 8; ++j) Vt[scol + j][srow] = vv[j];
        }
        // prefetch next tile toward the WGP while we compute on this one
        if (jt + 32 < SEQ) {
            __builtin_prefetch(Kb + (size_t)(jt + 32 + srow) * INNER + scol, 0, 1);
            __builtin_prefetch(Vb + (size_t)(jt + 32 + srow) * INNER + scol, 0, 1);
        }
        asm volatile("s_wait_asynccnt 0x0" ::: "memory");
        __syncthreads();   // tile visible to all waves

        // ---- S = Q @ K^T : two 16x16 score tiles from LDS ----
        half16 kb00 = lds_b_frag(&Kt[0][0], 0,  KT_LD, 0,  lane);
        half16 kb01 = lds_b_frag(&Kt[0][0], 0,  KT_LD, 32, lane);
        half16 kb10 = lds_b_frag(&Kt[0][0], 16, KT_LD, 0,  lane);
        half16 kb11 = lds_b_frag(&Kt[0][0], 16, KT_LD, 32, lane);
        float8 s0 = wmma_f16(qa0, kb00, z);
        s0        = wmma_f16(qa1, kb01, s0);
        float8 s1 = wmma_f16(qa0, kb10, z);
        s1        = wmma_f16(qa1, kb11, s1);

        // ---- online softmax across the 32 new columns ----
        float corr[8];
#pragma unroll
        for (int i = 0; i < 8; ++i) {
            float a   = s0[i] * SCALE_F;
            float bsc = s1[i] * SCALE_F;
            float v = fmaxf(a, bsc);
            v = fmaxf(v, __shfl_xor(v, 1));
            v = fmaxf(v, __shfl_xor(v, 2));
            v = fmaxf(v, __shfl_xor(v, 4));
            v = fmaxf(v, __shfl_xor(v, 8));
            float mn = fmaxf(m_i[i], v);
            corr[i] = __expf(m_i[i] - mn);
            float p0 = __expf(a - mn);
            float p1 = __expf(bsc - mn);
            s0[i] = p0;
            s1[i] = p1;
            float rs = p0 + p1;
            rs += __shfl_xor(rs, 1);
            rs += __shfl_xor(rs, 2);
            rs += __shfl_xor(rs, 4);
            rs += __shfl_xor(rs, 8);
            l_i[i] = l_i[i] * corr[i] + rs;
            m_i[i] = mn;
        }
#pragma unroll
        for (int i = 0; i < 8; ++i) {
            o0[i] *= corr[i]; o1[i] *= corr[i]; o2[i] *= corr[i]; o3[i] *= corr[i];
        }

        // ---- P (C-layout f32) -> per-wave LDS -> A fragment ----
#pragma unroll
        for (int i = 0; i < 8; ++i) {
            pb[(rr + i) * 32 + cc]      = (_Float16)s0[i];
            pb[(rr + i) * 32 + cc + 16] = (_Float16)s1[i];
        }
        __builtin_amdgcn_wave_barrier();
        half16 pa = load_a_f16(pb, 32, 0, 0, lane);
        __builtin_amdgcn_wave_barrier();

        // ---- O(16x64) += P(16x32) @ V(32x64), V fragments from transposed LDS ----
        half16 vb0 = lds_b_frag(&Vt[0][0], 0,  VT_LD, 0, lane);
        half16 vb1 = lds_b_frag(&Vt[0][0], 16, VT_LD, 0, lane);
        half16 vb2 = lds_b_frag(&Vt[0][0], 32, VT_LD, 0, lane);
        half16 vb3 = lds_b_frag(&Vt[0][0], 48, VT_LD, 0, lane);
        o0 = wmma_f16(pa, vb0, o0);
        o1 = wmma_f16(pa, vb1, o1);
        o2 = wmma_f16(pa, vb2, o2);
        o3 = wmma_f16(pa, vb3, o3);
    }

    // normalize and store (f16, interleaved [n, h*64+d])
#pragma unroll
    for (int i = 0; i < 8; ++i) {
        float inv = 1.0f / l_i[i];
        size_t row = (size_t)(qb + rr + i) * INNER;
        Ob[row + 0  + cc] = (_Float16)(o0[i] * inv);
        Ob[row + 16 + cc] = (_Float16)(o1[i] * inv);
        Ob[row + 32 + cc] = (_Float16)(o2[i] * inv);
        Ob[row + 48 + cc] = (_Float16)(o3[i] * inv);
    }
}

// ---- host side -------------------------------------------------------------
extern "C" void kernel_launch(void* const* d_in, const int* in_sizes, int n_in,
                              void* d_out, int out_size, void* d_ws, size_t ws_size,
                              hipStream_t stream) {
    (void)in_sizes; (void)n_in; (void)out_size; (void)ws_size;
    const float* x   = (const float*)d_in[0];
    const float* ctx = (const float*)d_in[1];
    const float* Wq  = (const float*)d_in[2];
    const float* Wk  = (const float*)d_in[3];
    const float* Wv  = (const float*)d_in[4];
    const float* Wo  = (const float*)d_in[5];
    const float* bo  = (const float*)d_in[6];
    float* out = (float*)d_out;

    const int BN = BATCH * SEQ;                // 8192
    _Float16* ws  = (_Float16*)d_ws;
    _Float16* Wqh = ws;                        // 512*512
    _Float16* Wkh = Wqh + INNER * QDIM;        // 512*768
    _Float16* Wvh = Wkh + INNER * CDIM;        // 512*768
    _Float16* Woh = Wvh + INNER * CDIM;        // 512*512
    _Float16* Qh  = Woh + QDIM * INNER;        // 8192*512
    _Float16* Kh  = Qh + (size_t)BN * INNER;
    _Float16* Vh  = Kh + (size_t)BN * INNER;
    _Float16* Oh  = Vh + (size_t)BN * INNER;

    cvt_f16<<<(INNER * QDIM + 255) / 256, 256, 0, stream>>>(Wq, Wqh, INNER * QDIM);
    cvt_f16<<<(INNER * CDIM + 255) / 256, 256, 0, stream>>>(Wk, Wkh, INNER * CDIM);
    cvt_f16<<<(INNER * CDIM + 255) / 256, 256, 0, stream>>>(Wv, Wvh, INNER * CDIM);
    cvt_f16<<<(QDIM * INNER + 255) / 256, 256, 0, stream>>>(Wo, Woh, QDIM * INNER);

    gemm_xw_f16<<<dim3(BN / 32, INNER / 128), 128, 0, stream>>>(x,   Wqh, Qh, BN, QDIM, INNER);
    gemm_xw_f16<<<dim3(BN / 32, INNER / 128), 128, 0, stream>>>(ctx, Wkh, Kh, BN, CDIM, INNER);
    gemm_xw_f16<<<dim3(BN / 32, INNER / 128), 128, 0, stream>>>(ctx, Wvh, Vh, BN, CDIM, INNER);

    attn_flash<<<dim3(SEQ / 128, HEADS, BATCH), 256, 0, stream>>>(Qh, Kh, Vh, Oh);

    gemm_ow_f32<<<dim3(BN / 32, QDIM / 128), 128, 0, stream>>>(Oh, Woh, bo, out, BN, INNER, QDIM);
}
